// UMambaBlock_21062519620110
// MI455X (gfx1250) — compile-verified, matrix operand order checked by reading, not run on hardware
//
#include <hip/hip_runtime.h>
#include <hip/hip_bf16.h>
#include <math.h>

// ---------------------------------------------------------------------------
// UMamba block for MI455X (gfx1250, wave32, WMMA 16x16x32 f16 -> f32 accum)
// A/B fragments are fed with contiguous 16B loads by exploiting the CDNA5
// 16-bit operand layout: lane elements j=0..7 -> K=8h..8h+7, j=8..15 ->
// K=16+8h..16+8h+7 (ISA 7.12.2), i.e. two aligned 16-byte runs per lane.
// ---------------------------------------------------------------------------
#define BB  16
#define CC  64
#define HH  64
#define WW  64
#define LL  (HH * WW)          // 4096
#define DI  128                // D_INNER
#define DSN 16                 // D_STATE
#define DTR 4                  // DT_RANK
#define NROWS (BB * LL)        // 65536
#define SLABP 72               // padded channel-minor row stride (f16)

typedef __attribute__((ext_vector_type(16))) _Float16 v16h;
typedef __attribute__((ext_vector_type(8)))  float    v8f;

union v16hu { v16h v; _Float16 e[16]; int4 q[2]; };
union v8fu  { v8f  v; float    e[8];  };
union pk8   { int4 q; _Float16 e[8]; };

#define WMMA(A, B, Cacc) \
  __builtin_amdgcn_wmma_f32_16x16x32_f16(false, (A), false, (B), (short)0, (Cacc), false, false)

// K index of element j for lane-half `half` (pack-side only)
__device__ __forceinline__ int k_of(int j, int half) {
  int v = j >> 1;
  int base = (v < 4) ? (2 * v) : (16 + 2 * (v - 4));
  return base + 8 * half + (j & 1);
}

// A fragment from a K-minor f16 row in LDS: two aligned 16B loads
__device__ __forceinline__ v16h a_from_lds(const _Float16* row, int kc, int half) {
  v16hu a;
  a.q[0] = *(const int4*)(row + kc * 32 + 8 * half);
  a.q[1] = *(const int4*)(row + kc * 32 + 16 + 8 * half);
  return a.v;
}

// A fragment from a K-contiguous f32 row in global: four float4 loads + cvt
__device__ __forceinline__ v16h a_from_f32row(const float* row, int kc, int half) {
  v16hu a;
  const float* p0 = row + kc * 32 + 8 * half;
  const float* p1 = row + kc * 32 + 16 + 8 * half;
  #pragma unroll
  for (int t = 0; t < 2; ++t) {
    float4 f = *(const float4*)(p0 + 4 * t);
    a.e[t * 4 + 0] = (_Float16)f.x; a.e[t * 4 + 1] = (_Float16)f.y;
    a.e[t * 4 + 2] = (_Float16)f.z; a.e[t * 4 + 3] = (_Float16)f.w;
  }
  #pragma unroll
  for (int t = 0; t < 2; ++t) {
    float4 f = *(const float4*)(p1 + 4 * t);
    a.e[8 + t * 4 + 0] = (_Float16)f.x; a.e[8 + t * 4 + 1] = (_Float16)f.y;
    a.e[8 + t * 4 + 2] = (_Float16)f.z; a.e[8 + t * 4 + 3] = (_Float16)f.w;
  }
  return a.v;
}

// ---------------------------------------------------------------------------
// Weight packing: fp32 -> f16 per-lane WMMA B fragments (one 32B load/lane).
// ---------------------------------------------------------------------------
__global__ void pack_conv_w(const float* __restrict__ w, _Float16* __restrict__ wp) {
  int t = blockIdx.x * blockDim.x + threadIdx.x;       // 9*2*4*32 = 2304
  if (t >= 9 * 2 * 4 * 32) return;
  int lane = t & 31, r = t >> 5;
  int ct = r & 3;  r >>= 2;
  int kc = r & 1;  r >>= 1;
  int s  = r;
  int dy = s / 3, dx = s % 3;
  int half = lane >> 4, n = lane & 15;
  int co = ct * 16 + n;
  for (int j = 0; j < 16; ++j) {
    int ci = kc * 32 + k_of(j, half);
    wp[(size_t)t * 16 + j] = (_Float16)w[((co * CC + ci) * 3 + dy) * 3 + dx];
  }
}

__global__ void pack_inproj(const float* __restrict__ w, _Float16* __restrict__ wp) {
  int t = blockIdx.x * blockDim.x + threadIdx.x;       // 2*16*32 = 1024
  if (t >= 2 * 16 * 32) return;
  int lane = t & 31, r = t >> 5;
  int ct = r & 15, kc = r >> 4;
  int half = lane >> 4, n = lane & 15;
  for (int j = 0; j < 16; ++j) {
    int k = kc * 32 + k_of(j, half);
    wp[(size_t)t * 16 + j] = (_Float16)w[k * 256 + ct * 16 + n];
  }
}

__global__ void pack_xproj(const float* __restrict__ w, _Float16* __restrict__ wp) {
  int t = blockIdx.x * blockDim.x + threadIdx.x;       // 4*3*32 = 384
  if (t >= 4 * 3 * 32) return;
  int lane = t & 31, r = t >> 5;
  int ct = r % 3, kc = r / 3;
  int half = lane >> 4, n = lane & 15;
  for (int j = 0; j < 16; ++j) {
    int k = kc * 32 + k_of(j, half);
    int col = ct * 16 + n;
    float v = (col < DTR + 2 * DSN) ? w[k * (DTR + 2 * DSN) + col] : 0.0f;
    wp[(size_t)t * 16 + j] = (_Float16)v;
  }
}

__global__ void pack_outproj(const float* __restrict__ w, _Float16* __restrict__ wp) {
  int t = blockIdx.x * blockDim.x + threadIdx.x;       // 4*4*32 = 512
  if (t >= 4 * 4 * 32) return;
  int lane = t & 31, r = t >> 5;
  int ct = r & 3, kc = r >> 2;
  int half = lane >> 4, n = lane & 15;
  for (int j = 0; j < 16; ++j) {
    int k = kc * 32 + k_of(j, half);
    wp[(size_t)t * 16 + j] = (_Float16)w[k * CC + ct * 16 + n];
  }
}

// ---------------------------------------------------------------------------
// 3x3 conv + leaky ReLU: 9 shifted channel-mix GEMMs, WMMA f16.
// Halo slab staged channel-minor [54 pos][72 ci] f16; octet staging with
// ds_store_b128; each A fragment is two ds_load_b128.
// ---------------------------------------------------------------------------
__global__ void __launch_bounds__(32)
conv3x3_lrelu(const float* __restrict__ in, const _Float16* __restrict__ wp,
              const float* __restrict__ bias, float* __restrict__ out) {
  __shared__ _Float16 lds[54 * SLABP];                 // 7776 B
  int blk = blockIdx.x;                                // b*256 + h*4 + wt
  int wt = blk & 3, h = (blk >> 2) & (HH - 1), b = blk >> 8;
  int w0 = wt * 16;
  int lane = threadIdx.x;

  // stage: unit = (pos, ci-octet); one 16B LDS store per unit
  for (int i = lane; i < 54 * 8; i += 32) {
    int pos = i >> 3, oct = i & 7;
    int dy = pos / 18, c = pos % 18;
    int hh = h + dy - 1, ww = w0 + c - 1;
    pk8 pk;
    if ((unsigned)hh < (unsigned)HH && (unsigned)ww < (unsigned)WW) {
      const float* gp = in + ((size_t)b * CC + oct * 8) * LL + hh * WW + ww;
      #pragma unroll
      for (int q = 0; q < 8; ++q) pk.e[q] = (_Float16)gp[(size_t)q * LL];
    } else {
      #pragma unroll
      for (int q = 0; q < 8; ++q) pk.e[q] = (_Float16)0.0f;
    }
    *(int4*)&lds[pos * SLABP + oct * 8] = pk.q;
  }
  if (wt < 3)
    __builtin_prefetch(&in[(((size_t)b * CC) * HH + h) * WW + w0 + 16], 0, 0);
  __syncthreads();

  int half = lane >> 4, m = lane & 15, n = m;
  v8fu acc[4];
  acc[0].v = (v8f)0.f; acc[1].v = (v8f)0.f; acc[2].v = (v8f)0.f; acc[3].v = (v8f)0.f;

  for (int s = 0; s < 9; ++s) {
    int dy = s / 3, dx = s % 3;
    const _Float16* row = &lds[(dy * 18 + dx + m) * SLABP];
    #pragma unroll
    for (int kc = 0; kc < 2; ++kc) {
      v16h a = a_from_lds(row, kc, half);
      const _Float16* bp = wp + (size_t)(((s * 2 + kc) * 4) * 32 + lane) * 16;
      v16h b0 = *(const v16h*)(bp);
      v16h b1 = *(const v16h*)(bp + 512);
      v16h b2 = *(const v16h*)(bp + 1024);
      v16h b3 = *(const v16h*)(bp + 1536);
      acc[0].v = WMMA(a, b0, acc[0].v);
      acc[1].v = WMMA(a, b1, acc[1].v);
      acc[2].v = WMMA(a, b2, acc[2].v);
      acc[3].v = WMMA(a, b3, acc[3].v);
    }
  }
  for (int ct = 0; ct < 4; ++ct) {
    int co = ct * 16 + n;
    float bv = bias[co];
    size_t obase = (((size_t)b * CC + co) * HH + h) * WW + w0 + half * 8;
    #pragma unroll
    for (int v = 0; v < 8; ++v) {
      float y = acc[ct].e[v] + bv;
      out[obase + v] = (y >= 0.0f) ? y : 0.01f * y;
    }
  }
}

// ---------------------------------------------------------------------------
// in_proj: [B*L,64] @ [64,256].  Activation tile staged channel-minor in LDS;
// xz epilogue goes through LDS so global stores are coalesced b128.
// ---------------------------------------------------------------------------
__global__ void __launch_bounds__(32)
in_proj_wmma(const float* __restrict__ y2, const _Float16* __restrict__ ipp,
             float* __restrict__ xin, float* __restrict__ zbuf) {
  __shared__ _Float16 tl[16 * SLABP];                  // 2304 B
  __shared__ float    ot[16 * 256];                    // 16 KB
  int blk = blockIdx.x;                                // b*256 + tile
  int tile = blk & 255, b = blk >> 8;
  int l0 = tile * 16;
  int lane = threadIdx.x, half = lane >> 4, m = lane & 15, n = m;

  // stage 16 rows x 64 ci, channel-minor
  for (int i = lane; i < 16 * 8; i += 32) {
    int m2 = i >> 3, oct = i & 7;
    pk8 pk;
    const float* gp = y2 + ((size_t)b * CC + oct * 8) * LL + l0 + m2;
    #pragma unroll
    for (int q = 0; q < 8; ++q) pk.e[q] = (_Float16)gp[(size_t)q * LL];
    *(int4*)&tl[m2 * SLABP + oct * 8] = pk.q;
  }
  __syncthreads();

  const _Float16* row = &tl[m * SLABP];
  v16h a0 = a_from_lds(row, 0, half);
  v16h a1 = a_from_lds(row, 1, half);

  for (int ct = 0; ct < 16; ++ct) {
    v16h b0 = *(const v16h*)(ipp + (size_t)((ct)      * 32 + lane) * 16);
    v16h b1 = *(const v16h*)(ipp + (size_t)((16 + ct) * 32 + lane) * 16);
    v8fu acc; acc.v = (v8f)0.f;
    acc.v = WMMA(a0, b0, acc.v);
    acc.v = WMMA(a1, b1, acc.v);
    int col = ct * 16 + n;
    #pragma unroll
    for (int v = 0; v < 8; ++v)
      ot[(v + half * 8) * 256 + col] = acc.e[v];
  }
  __syncthreads();

  // coalesced writeout: x half and z half, float4 per unit
  size_t rbase = (size_t)b * LL + l0;
  for (int u = lane; u < 16 * 64; u += 32) {
    int m2 = u >> 6, c4 = u & 63;
    float4 val = *(const float4*)&ot[m2 * 256 + c4 * 4];
    if (c4 < 32)
      *(float4*)&xin[(rbase + m2) * DI + c4 * 4] = val;
    else
      *(float4*)&zbuf[(rbase + m2) * DI + (c4 - 32) * 4] = val;
  }
}

// ---------------------------------------------------------------------------
// depthwise causal conv1d (k=4) + SiLU
// ---------------------------------------------------------------------------
__global__ void conv1d_silu(const float* __restrict__ xin, const float* __restrict__ w,
                            const float* __restrict__ bias, float* __restrict__ xs) {
  size_t idx = (size_t)blockIdx.x * blockDim.x + threadIdx.x;
  if (idx >= (size_t)NROWS * DI) return;
  int d = (int)(idx & (DI - 1));
  size_t bl = idx >> 7;
  int l = (int)(bl & (LL - 1));
  float acc = bias[d];
  #pragma unroll
  for (int t = 0; t < 4; ++t) {
    int ll = l - 3 + t;
    if (ll >= 0) acc += w[d * 4 + t] * xin[(bl - (size_t)(3 - t)) * DI + d];
  }
  xs[idx] = acc / (1.0f + __expf(-acc));
}

// ---------------------------------------------------------------------------
// x_proj: [B*L,128] @ [128,48(pad 36)] -> dbc[row][48]
// ---------------------------------------------------------------------------
__global__ void __launch_bounds__(32)
xproj_wmma(const float* __restrict__ xs, const _Float16* __restrict__ xpp,
           float* __restrict__ dbc) {
  int r0 = blockIdx.x * 16;
  int lane = threadIdx.x, half = lane >> 4, m = lane & 15, n = m;

  const float* src = xs + (size_t)(r0 + m) * DI;
  v16h a[4];
  #pragma unroll
  for (int kc = 0; kc < 4; ++kc) a[kc] = a_from_f32row(src, kc, half);

  for (int ct = 0; ct < 3; ++ct) {
    v8fu acc; acc.v = (v8f)0.f;
    #pragma unroll
    for (int kc = 0; kc < 4; ++kc) {
      v16h bv = *(const v16h*)(xpp + (size_t)((kc * 3 + ct) * 32 + lane) * 16);
      acc.v = WMMA(a[kc], bv, acc.v);
    }
    #pragma unroll
    for (int v = 0; v < 8; ++v)
      dbc[(size_t)(r0 + v + half * 8) * 48 + ct * 16 + n] = acc.e[v];
  }
}

// ---------------------------------------------------------------------------
// delta = softplus(dt @ dt_proj_w + dt_proj_b)
// ---------------------------------------------------------------------------
__global__ void delta_kernel(const float* __restrict__ dbc, const float* __restrict__ dtw,
                             const float* __restrict__ dtb, float* __restrict__ delta) {
  size_t idx = (size_t)blockIdx.x * blockDim.x + threadIdx.x;
  if (idx >= (size_t)NROWS * DI) return;
  int d = (int)(idx & (DI - 1));
  size_t row = idx >> 7;
  float4 dt = *(const float4*)&dbc[row * 48];
  float acc = dtb[d] + dt.x * dtw[0 * DI + d] + dt.y * dtw[1 * DI + d]
                     + dt.z * dtw[2 * DI + d] + dt.w * dtw[3 * DI + d];
  delta[idx] = (acc > 20.0f) ? acc : log1pf(__expf(acc));
}

// ---------------------------------------------------------------------------
// Selective scan, lane = state n (16 per channel, 2 channels per wave),
// software-pipelined loads, shfl_xor state reduction, fused D/silu(z) gate.
// ---------------------------------------------------------------------------
__global__ void __launch_bounds__(128)
scan_kernel(const float* __restrict__ delta, const float* __restrict__ xs,
            const float* __restrict__ dbc, const float* __restrict__ zbuf,
            const float* __restrict__ A_log, const float* __restrict__ Dvec,
            float* __restrict__ yg) {
  int gw = (blockIdx.x * blockDim.x + threadIdx.x) >> 5;   // 0..1023
  int lane = threadIdx.x & 31;
  int half = lane >> 4, n = lane & 15;
  int b = gw >> 6;
  int d = (gw & 63) * 2 + half;

  float A_dn = -__expf(A_log[d * DSN + n]);
  float Dd = Dvec[d];
  float h = 0.0f;
  size_t rbase = (size_t)b * LL;

  // prologue loads for l = 0
  float dv = delta[rbase * DI + d];
  float xv = xs[rbase * DI + d];
  float Bv = dbc[rbase * 48 + DTR + n];
  float Cv = dbc[rbase * 48 + DTR + DSN + n];
  float zv = zbuf[rbase * DI + d];

  for (int l = 0; l < LL; ++l) {
    float dv_n = 0.f, xv_n = 0.f, Bv_n = 0.f, Cv_n = 0.f, zv_n = 0.f;
    if (l + 1 < LL) {                                   // prefetch next step
      size_t r2 = rbase + l + 1;
      dv_n = delta[r2 * DI + d];
      xv_n = xs[r2 * DI + d];
      Bv_n = dbc[r2 * 48 + DTR + n];
      Cv_n = dbc[r2 * 48 + DTR + DSN + n];
      zv_n = zbuf[r2 * DI + d];
    }
    h = __expf(dv * A_dn) * h + (dv * xv) * Bv;
    float p = h * Cv;
    p += __shfl_xor(p, 1);
    p += __shfl_xor(p, 2);
    p += __shfl_xor(p, 4);
    p += __shfl_xor(p, 8);
    if (n == 0) {
      float y = p + xv * Dd;
      yg[(rbase + l) * DI + d] = y * (zv / (1.0f + __expf(-zv)));
    }
    dv = dv_n; xv = xv_n; Bv = Bv_n; Cv = Cv_n; zv = zv_n;
  }
}

// ---------------------------------------------------------------------------
// out_proj: [B*L,128] @ [128,64] -> out[b][co][l]  (contiguous stores)
// ---------------------------------------------------------------------------
__global__ void __launch_bounds__(32)
out_proj_wmma(const float* __restrict__ yg, const _Float16* __restrict__ opp,
              float* __restrict__ out) {
  int r0 = blockIdx.x * 16;
  int lane = threadIdx.x, half = lane >> 4, m = lane & 15, n = m;
  int b = r0 >> 12, l0 = r0 & (LL - 1);

  const float* src = yg + (size_t)(r0 + m) * DI;
  v16h a[4];
  #pragma unroll
  for (int kc = 0; kc < 4; ++kc) a[kc] = a_from_f32row(src, kc, half);

  for (int ct = 0; ct < 4; ++ct) {
    v8fu acc; acc.v = (v8f)0.f;
    #pragma unroll
    for (int kc = 0; kc < 4; ++kc) {
      v16h bv = *(const v16h*)(opp + (size_t)((kc * 4 + ct) * 32 + lane) * 16);
      acc.v = WMMA(a[kc], bv, acc.v);
    }
    int co = ct * 16 + n;
    size_t obase = ((size_t)b * CC + co) * LL + l0 + half * 8;
    #pragma unroll
    for (int v = 0; v < 8; ++v)
      out[obase + v] = acc.e[v];
  }
}

// ---------------------------------------------------------------------------
extern "C" void kernel_launch(void* const* d_in, const int* in_sizes, int n_in,
                              void* d_out, int out_size, void* d_ws, size_t ws_size,
                              hipStream_t stream) {
  (void)in_sizes; (void)n_in; (void)out_size; (void)ws_size;

  const float* x        = (const float*)d_in[0];
  const float* conv_w1  = (const float*)d_in[1];
  const float* conv_b1  = (const float*)d_in[2];
  const float* conv_w2  = (const float*)d_in[3];
  const float* conv_b2  = (const float*)d_in[4];
  const float* in_w     = (const float*)d_in[5];
  const float* c1d_w    = (const float*)d_in[6];
  const float* c1d_b    = (const float*)d_in[7];
  const float* xp_w     = (const float*)d_in[8];
  const float* dt_w     = (const float*)d_in[9];
  const float* dt_b     = (const float*)d_in[10];
  const float* A_log    = (const float*)d_in[11];
  const float* Dvec     = (const float*)d_in[12];
  const float* op_w     = (const float*)d_in[13];
  float* out = (float*)d_out;

  float* p = (float*)d_ws;
  float* y1     = p;  p += (size_t)BB * CC * LL;
  float* y2     = p;  p += (size_t)BB * CC * LL;
  float* xin    = p;  p += (size_t)NROWS * DI;
  float* zbuf   = p;  p += (size_t)NROWS * DI;
  float* xsb    = p;  p += (size_t)NROWS * DI;
  float* dbc    = p;  p += (size_t)NROWS * 48;
  float* deltab = p;  p += (size_t)NROWS * DI;
  float* yg     = p;  p += (size_t)NROWS * DI;
  _Float16* w1p = (_Float16*)p;
  _Float16* w2p = w1p + 9 * 2 * 4 * 32 * 16;
  _Float16* ipp = w2p + 9 * 2 * 4 * 32 * 16;
  _Float16* xpp = ipp + 2 * 16 * 32 * 16;
  _Float16* opp = xpp + 4 * 3 * 32 * 16;

  pack_conv_w<<<9, 256, 0, stream>>>(conv_w1, w1p);
  pack_conv_w<<<9, 256, 0, stream>>>(conv_w2, w2p);
  pack_inproj<<<4, 256, 0, stream>>>(in_w, ipp);
  pack_xproj <<<2, 256, 0, stream>>>(xp_w, xpp);
  pack_outproj<<<2, 256, 0, stream>>>(op_w, opp);

  conv3x3_lrelu<<<BB * HH * 4, 32, 0, stream>>>(x,  w1p, conv_b1, y1);
  conv3x3_lrelu<<<BB * HH * 4, 32, 0, stream>>>(y1, w2p, conv_b2, y2);

  in_proj_wmma<<<BB * (LL / 16), 32, 0, stream>>>(y2, ipp, xin, zbuf);

  size_t nel = (size_t)NROWS * DI;
  conv1d_silu<<<(unsigned)((nel + 255) / 256), 256, 0, stream>>>(xin, c1d_w, c1d_b, xsb);

  xproj_wmma<<<NROWS / 16, 32, 0, stream>>>(xsb, xpp, dbc);
  delta_kernel<<<(unsigned)((nel + 255) / 256), 256, 0, stream>>>(dbc, dt_w, dt_b, deltab);

  scan_kernel<<<(BB * DI / 2) * 32 / 128, 128, 0, stream>>>(deltab, xsb, dbc, zbuf,
                                                            A_log, Dvec, yg);

  out_proj_wmma<<<NROWS / 16, 32, 0, stream>>>(yg, opp, out);
}
